// GPT2_67654324846742
// MI455X (gfx1250) — matrix operand order repaired and checked by reading, not compiled, not run on hardware
//
#include <hip/hip_runtime.h>

// ---------------- types ----------------
typedef __attribute__((ext_vector_type(16))) _Float16 v16h;
typedef __attribute__((ext_vector_type(8)))  _Float16 v8h;
typedef __attribute__((ext_vector_type(4)))  _Float16 v4h;
typedef __attribute__((ext_vector_type(8)))  float    v8f;

// ---------------- model dims ----------------
#define LNUM 12
#define EE   1024
#define NH   16
#define TT   512
#define BBATCH 8
#define VV   512
#define FF   4096
#define DD   64
#define MM   (BBATCH*TT)   // 4096 rows

// GEMM tiling
#define BM 128
#define BN 64
#define BK 64
#define LPAD 8            // LDS row pad: +8 halves keeps 16B alignment, skews banks
#define ASTR (BK + LPAD)  // 72 halves = 144B
#define BSTR (BN + LPAD)  // 72 halves = 144B

// epilogue modes
#define MODE_F32   0   // out f32 (+bias if non-null)
#define MODE_QKV   1   // out f16, split heads: [B,H,T,D]
#define MODE_GELU  2   // out f16, exact gelu
#define MODE_RES   3   // out f32 = acc + bias + res (residual add)

// ---------------- helpers ----------------
__device__ inline v16h comb8(const _Float16* p0, const _Float16* p1) {
    v8h a = *(const v8h*)p0;
    v8h b = *(const v8h*)p1;
    v16h r;
#pragma unroll
    for (int i = 0; i < 8; ++i) { r[i] = a[i]; r[i + 8] = b[i]; }
    return r;
}

__device__ inline float red_max16(float x) {
#pragma unroll
    for (int m = 1; m < 16; m <<= 1) x = fmaxf(x, __shfl_xor(x, m, 32));
    return x;
}
__device__ inline float red_sum16(float x) {
#pragma unroll
    for (int m = 1; m < 16; m <<= 1) x += __shfl_xor(x, m, 32);
    return x;
}

// ---------------- embedding: h = tok_emb[x] + pos_emb ----------------
__global__ __launch_bounds__(256) void embed_kernel(
    const int* __restrict__ x, const float* __restrict__ tok,
    const float* __restrict__ pos, float* __restrict__ h)
{
    int row = blockIdx.x;                 // b*T + t
    int t   = row % TT;
    int tokid = x[row];
    int e = threadIdx.x * 4;
    float4 tv = *(const float4*)(tok + (size_t)tokid * EE + e);
    float4 pv = *(const float4*)(pos + (size_t)t * EE + e);
    float4 o; o.x = tv.x + pv.x; o.y = tv.y + pv.y; o.z = tv.z + pv.z; o.w = tv.w + pv.w;
    *(float4*)(h + (size_t)row * EE + e) = o;
}

// ---------------- layernorm (fp32 in, f16 out) ----------------
__global__ __launch_bounds__(256) void ln_kernel(
    const float* __restrict__ xin, const float* __restrict__ g,
    const float* __restrict__ b, _Float16* __restrict__ out)
{
    int row = blockIdx.x;
    int tid = threadIdx.x, lane = tid & 31, wid = tid >> 5;
    const float* xr = xin + (size_t)row * EE;
    float4 xv = *(const float4*)(xr + tid * 4);
    float s  = xv.x + xv.y + xv.z + xv.w;
    float sq = xv.x * xv.x + xv.y * xv.y + xv.z * xv.z + xv.w * xv.w;
#pragma unroll
    for (int m = 1; m < 32; m <<= 1) { s += __shfl_xor(s, m, 32); sq += __shfl_xor(sq, m, 32); }
    __shared__ float ws1[8], ws2[8];
    __shared__ float mu_s, rs_s;
    if (lane == 0) { ws1[wid] = s; ws2[wid] = sq; }
    __syncthreads();
    if (tid == 0) {
        float S = 0.f, Q = 0.f;
#pragma unroll
        for (int i = 0; i < 8; ++i) { S += ws1[i]; Q += ws2[i]; }
        float mu = S * (1.0f / EE);
        float var = Q * (1.0f / EE) - mu * mu;
        mu_s = mu; rs_s = rsqrtf(var + 1e-5f);
    }
    __syncthreads();
    float mu = mu_s, rstd = rs_s;
    int e = tid * 4;
    v4h o;
    o[0] = (_Float16)((xv.x - mu) * rstd * g[e + 0] + b[e + 0]);
    o[1] = (_Float16)((xv.y - mu) * rstd * g[e + 1] + b[e + 1]);
    o[2] = (_Float16)((xv.z - mu) * rstd * g[e + 2] + b[e + 2]);
    o[3] = (_Float16)((xv.w - mu) * rstd * g[e + 3] + b[e + 3]);
    *(v4h*)(out + (size_t)row * EE + e) = o;
}

// ---------------- WMMA GEMM: C[M,N] = A_f16[M,K] @ B_f32[K,N] + epilogue ----------------
// BM=128 x BN=64 block tile, BK=64 K-steps, register-staged double-buffered LDS
// pipeline: issue next-tile global loads -> 8 WMMAs on current tile -> spill regs
// to other LDS buffer -> one barrier per iteration.
__global__ __launch_bounds__(256) void gemm_wmma(
    const _Float16* __restrict__ A, const float* __restrict__ Bw,
    const float* __restrict__ bias, const float* __restrict__ res,
    float* __restrict__ outF, _Float16* __restrict__ outH,
    int M, int N, int K, int mode)
{
    __shared__ alignas(16) _Float16 As[2][BM][ASTR];   // 2*128*72*2 = 36864 B
    __shared__ alignas(16) _Float16 Bs[2][BK][BSTR];   // 2*64*72*2  = 18432 B

    const int tid  = threadIdx.x;
    const int lane = tid & 31, wid = tid >> 5;
    const int wm = wid & 3, wn = wid >> 2;        // 4 x 2 wave grid
    const int m0 = blockIdx.y * BM, n0 = blockIdx.x * BN;

    v8f acc[2][2];
#pragma unroll
    for (int i = 0; i < 2; ++i)
#pragma unroll
        for (int j = 0; j < 2; ++j) acc[i][j] = (v8f)0.0f;

    const int kA = (lane < 16) ? 0 : 8;           // A-fragment K base split

    // per-thread staging coordinates
    const int ar_row[1] = {0}; (void)ar_row;

    // ---- stage tile 0 directly into LDS buffer 0 ----
#pragma unroll
    for (int i = 0; i < 4; ++i) {                 // A: 128x64 halves, 1024 v8h
        int idx = tid + 256 * i;
        int r = idx >> 3, seg = idx & 7;
        v8h av = *(const v8h*)(A + (size_t)(m0 + r) * K + seg * 8);
        *(v8h*)&As[0][r][seg * 8] = av;
    }
#pragma unroll
    for (int i = 0; i < 4; ++i) {                 // B: 64x64 floats, 1024 float4
        int idx = tid + 256 * i;
        int r = idx >> 4, c4 = idx & 15;
        float4 bv = *(const float4*)(Bw + (size_t)r * N + n0 + c4 * 4);
        v4h hv;
        hv[0] = (_Float16)bv.x; hv[1] = (_Float16)bv.y;
        hv[2] = (_Float16)bv.z; hv[3] = (_Float16)bv.w;
        *(v4h*)&Bs[0][r][c4 * 4] = hv;
    }
    __syncthreads();

    int cur = 0;
    for (int kb = 0; kb < K; kb += BK) {
        const bool hasNext = (kb + BK) < K;

        // ---- issue next tile's global loads into registers ----
        v8h ar[4]; float4 br[4];
        if (hasNext) {
#pragma unroll
            for (int i = 0; i < 4; ++i) {
                int idx = tid + 256 * i;
                int r = idx >> 3, seg = idx & 7;
                ar[i] = *(const v8h*)(A + (size_t)(m0 + r) * K + kb + BK + seg * 8);
            }
#pragma unroll
            for (int i = 0; i < 4; ++i) {
                int idx = tid + 256 * i;
                int r = idx >> 4, c4 = idx & 15;
                br[i] = *(const float4*)(Bw + (size_t)(kb + BK + r) * N + n0 + c4 * 4);
            }
            // prefetch tile after next into L2/WGP caches (global_prefetch_b8)
            if (kb + 2 * BK < K) {
                int r = tid >> 1;                                     // 128 rows, 2 thr/row
                __builtin_prefetch(A + (size_t)(m0 + r) * K + kb + 2 * BK + (tid & 1) * 32, 0, 1);
                if (tid < 128) {
                    int rb = tid >> 1;                                // 64 rows, 2 thr/row
                    __builtin_prefetch(Bw + (size_t)(kb + 2 * BK + rb) * N + n0 + (tid & 1) * 32, 0, 1);
                }
            }
        }

        // ---- compute on current LDS buffer: 2 K-steps of 32, 8 WMMAs ----
#pragma unroll
        for (int ks = 0; ks < BK; ks += 32) {
            v16h af[2], bf[2];
#pragma unroll
            for (int mt = 0; mt < 2; ++mt) {
                int mr = wm * 32 + mt * 16 + (lane & 15);
                af[mt] = comb8(&As[cur][mr][ks + kA], &As[cur][mr][ks + kA + 16]);
            }
#pragma unroll
            for (int nt = 0; nt < 2; ++nt) {
                int nc = wn * 32 + nt * 16;
#pragma unroll
                for (int s = 0; s < 8; ++s) {
                    bf[nt][s]     = Bs[cur][ks + kA + s][nc + (lane & 15)];
                    bf[nt][s + 8] = Bs[cur][ks + kA + 16 + s][nc + (lane & 15)];
                }
            }
#pragma unroll
            for (int mt = 0; mt < 2; ++mt)
#pragma unroll
                for (int nt = 0; nt < 2; ++nt)
                    acc[mt][nt] = __builtin_amdgcn_wmma_f32_16x16x32_f16(
                        false, af[mt], false, bf[nt], (short)0, acc[mt][nt], false, false);
        }

        // ---- spill staged registers into the other LDS buffer ----
        if (hasNext) {
            int nxt = cur ^ 1;
#pragma unroll
            for (int i = 0; i < 4; ++i) {
                int idx = tid + 256 * i;
                int r = idx >> 3, seg = idx & 7;
                *(v8h*)&As[nxt][r][seg * 8] = ar[i];
            }
#pragma unroll
            for (int i = 0; i < 4; ++i) {
                int idx = tid + 256 * i;
                int r = idx >> 4, c4 = idx & 15;
                v4h hv;
                hv[0] = (_Float16)br[i].x; hv[1] = (_Float16)br[i].y;
                hv[2] = (_Float16)br[i].z; hv[3] = (_Float16)br[i].w;
                *(v4h*)&Bs[nxt][r][c4 * 4] = hv;
            }
        }
        __syncthreads();
        cur ^= 1;
    }

    // epilogue: element (v, lane): M = v + 8*(lane/16), N = lane%16 per 16x16 tile
#pragma unroll
    for (int mt = 0; mt < 2; ++mt)
#pragma unroll
        for (int nt = 0; nt < 2; ++nt)
#pragma unroll
            for (int v = 0; v < 8; ++v) {
                int m = m0 + wm * 32 + mt * 16 + v + 8 * (lane >> 4);
                int n = n0 + wn * 32 + nt * 16 + (lane & 15);
                float val = acc[mt][nt][v];
                if (bias) val += bias[n];
                if (mode == MODE_F32) {
                    outF[(size_t)m * N + n] = val;
                } else if (mode == MODE_QKV) {
                    int bi = m / TT, ti = m % TT, hi = n >> 6, di = n & 63;
                    outH[(((size_t)bi * NH + hi) * TT + ti) * DD + di] = (_Float16)val;
                } else if (mode == MODE_GELU) {
                    float gl = 0.5f * val * (1.0f + erff(val * 0.70710678118f));
                    outH[(size_t)m * N + n] = (_Float16)gl;
                } else { // MODE_RES
                    outF[(size_t)m * N + n] = val + res[(size_t)m * N + n];
                }
            }
}

// ---------------- fused causal flash attention ----------------
// grid: B*H*(T/16)/8 blocks, 8 waves/block; one wave owns one 16-query tile.
#define KTPAD 40   // 32 keys + 8 pad halves per LDS row
__global__ __launch_bounds__(256) void attn_kernel(
    const _Float16* __restrict__ q, const _Float16* __restrict__ k,
    const _Float16* __restrict__ v, _Float16* __restrict__ y)
{
    __shared__ alignas(16) _Float16 smem[8][64 * KTPAD + 16 * KTPAD];
    const int lane = threadIdx.x & 31, wid = threadIdx.x >> 5;
    _Float16* Kt = &smem[wid][0];              // [64 d][32 keys] transposed chunk
    _Float16* Pl = Kt + 64 * KTPAD;            // [16 q][32 keys] probabilities

    int tile = blockIdx.x * 8 + wid;           // 0 .. B*H*T/16-1
    int qt = tile & ((TT / 16) - 1);
    int bh = tile / (TT / 16);
    int hi = bh % NH, bi = bh / NH;
    int q0 = qt * 16;

    const _Float16* Qp = q + (size_t)bh * TT * DD;
    const _Float16* Kp = k + (size_t)bh * TT * DD;
    const _Float16* Vp = v + (size_t)bh * TT * DD;

    const int kA = (lane < 16) ? 0 : 8;

    // Q A-fragments (contraction over D=64 -> 2 fragments of K=32)
    const _Float16* Qr = Qp + (size_t)(q0 + (lane & 15)) * DD;
    v16h aq0 = comb8(Qr + kA,      Qr + kA + 16);
    v16h aq1 = comb8(Qr + 32 + kA, Qr + 48 + kA);

    float mrun[8], lrun[8];
    v8f o[4];
#pragma unroll
    for (int i = 0; i < 8; ++i) { mrun[i] = -1e30f; lrun[i] = 0.0f; }
#pragma unroll
    for (int i = 0; i < 4; ++i) o[i] = (v8f)0.0f;

    for (int j = 0; j < q0 + 16; j += 32) {
        // stage K chunk transposed: Kt[d][key_local]
        const _Float16* Kr = Kp + (size_t)(j + lane) * DD;
#pragma unroll
        for (int c = 0; c < 8; ++c) {
            v8h kv = *(const v8h*)(Kr + c * 8);
#pragma unroll
            for (int e = 0; e < 8; ++e) Kt[(c * 8 + e) * KTPAD + lane] = kv[e];
        }

        // S = (Q K^T) * scale, two 16-key tiles, causal mask, in C-layout
        float s[2][8];
#pragma unroll
        for (int kt = 0; kt < 2; ++kt) {
            v16h b0, b1;
#pragma unroll
            for (int s = 0; s < 8; ++s) {
                b0[s]     = Kt[(kA + s) * KTPAD + kt * 16 + (lane & 15)];
                b0[s + 8] = Kt[(kA + 16 + s) * KTPAD + kt * 16 + (lane & 15)];
                b1[s]     = Kt[(32 + kA + s) * KTPAD + kt * 16 + (lane & 15)];
                b1[s + 8] = Kt[(48 + kA + s) * KTPAD + kt * 16 + (lane & 15)];
            }
            v8f sv = (v8f)0.0f;
            sv = __builtin_amdgcn_wmma_f32_16x16x32_f16(false, aq0, false, b0, (short)0, sv, false, false);
            sv = __builtin_amdgcn_wmma_f32_16x16x32_f16(false, aq1, false, b1, (short)0, sv, false, false);
            int kg = j + kt * 16 + (lane & 15);
#pragma unroll
            for (int vv = 0; vv < 8; ++vv) {
                int qg = q0 + vv + 8 * (lane >> 4);
                float x = sv[vv] * 0.125f;        // 1/sqrt(64)
                s[kt][vv] = (kg > qg) ? -1e30f : x;
            }
        }

        // online softmax update per row
#pragma unroll
        for (int vv = 0; vv < 8; ++vv) {
            float rm = fmaxf(red_max16(s[0][vv]), red_max16(s[1][vv]));
            float mnew = fmaxf(mrun[vv], rm);
            float f = __expf(mrun[vv] - mnew);
            float p0 = __expf(s[0][vv] - mnew);
            float p1 = __expf(s[1][vv] - mnew);
            lrun[vv] = lrun[vv] * f + red_sum16(p0) + red_sum16(p1);
            mrun[vv] = mnew;
#pragma unroll
            for (int nt = 0; nt < 4; ++nt) o[nt][vv] *= f;
            int Ml = vv + 8 * (lane >> 4);
            Pl[Ml * KTPAD + (lane & 15)]      = (_Float16)p0;
            Pl[Ml * KTPAD + 16 + (lane & 15)] = (_Float16)p1;
        }

        // P (A-layout) and V (B-layout), O += P @ V
        v16h ap = comb8(&Pl[(lane & 15) * KTPAD + kA], &Pl[(lane & 15) * KTPAD + kA + 16]);
        const _Float16* Vr = Vp + (size_t)(j + lane) * DD;
#pragma unroll
        for (int nt = 0; nt < 4; ++nt) {
            v16h bv;
#pragma unroll
            for (int s = 0; s < 8; ++s) {
                bv[s]     = Vp[(size_t)(j + kA + s) * DD + nt * 16 + (lane & 15)];
                bv[s + 8] = Vp[(size_t)(j + kA + 16 + s) * DD + nt * 16 + (lane & 15)];
            }
            o[nt] = __builtin_amdgcn_wmma_f32_16x16x32_f16(false, ap, false, bv, (short)0, o[nt], false, false);
        }
    }

    // write combined-head output y[(b*T + t)*E + h*64 + d]
#pragma unroll
    for (int nt = 0; nt < 4; ++nt)
#pragma unroll
        for (int vv = 0; vv < 8; ++vv) {
            int Ml = vv + 8 * (lane >> 4);
            float val = o[nt][vv] / lrun[vv];
            y[((size_t)bi * TT + q0 + Ml) * EE + hi * DD + nt * 16 + (lane & 15)] = (_Float16)val;
        }
}

// ---------------- host launcher ----------------
extern "C" void kernel_launch(void* const* d_in, const int* in_sizes, int n_in,
                              void* d_out, int out_size, void* d_ws, size_t ws_size,
                              hipStream_t stream) {
    (void)in_sizes; (void)n_in; (void)out_size; (void)ws_size;
    const int*   xin  = (const int*)  d_in[0];
    // d_in[1] = targets (unused by forward)
    const float* tok  = (const float*)d_in[2];
    const float* pos  = (const float*)d_in[3];
    const float* ln1g = (const float*)d_in[4];
    const float* ln1b = (const float*)d_in[5];
    const float* Wq   = (const float*)d_in[6];
    const float* bq   = (const float*)d_in[7];
    const float* Wk   = (const float*)d_in[8];
    const float* bk   = (const float*)d_in[9];
    const float* Wv   = (const float*)d_in[10];
    const float* bv   = (const float*)d_in[11];
    const float* Wp   = (const float*)d_in[12];
    const float* bp   = (const float*)d_in[13];
    const float* ln2g = (const float*)d_in[14];
    const float* ln2b = (const float*)d_in[15];
    const float* Wf1  = (const float*)d_in[16];
    const float* bf1  = (const float*)d_in[17];
    const float* Wf2  = (const float*)d_in[18];
    const float* bf2  = (const float*)d_in[19];
    const float* lnfg = (const float*)d_in[20];
    const float* lnfb = (const float*)d_in[21];
    const float* headw= (const float*)d_in[22];

    char* w = (char*)d_ws;
    size_t off = 0;
    float*    h   = (float*)(w + off);    off += (size_t)MM * EE * 4;   // 16 MB
    _Float16* xn  = (_Float16*)(w + off); off += (size_t)MM * EE * 2;   //  8 MB
    _Float16* qh  = (_Float16*)(w + off); off += (size_t)MM * EE * 2;
    _Float16* kh  = (_Float16*)(w + off); off += (size_t)MM * EE * 2;
    _Float16* vh  = (_Float16*)(w + off); off += (size_t)MM * EE * 2;
    _Float16* yh  = (_Float16*)(w + off); off += (size_t)MM * EE * 2;
    _Float16* hid = (_Float16*)(w + off); off += (size_t)MM * FF * 2;   // 32 MB

    dim3 blk(256);
    dim3 gE(EE / BN, MM / BM);   // N=1024
    dim3 gF(FF / BN, MM / BM);   // N=4096
    dim3 gV(VV / BN, MM / BM);   // N=512

    embed_kernel<<<MM, blk, 0, stream>>>(xin, tok, pos, h);

    for (int l = 0; l < LNUM; ++l) {
        const size_t we = (size_t)l * EE * EE;
        const size_t wf = (size_t)l * EE * FF;
        ln_kernel<<<MM, blk, 0, stream>>>(h, ln1g + l * EE, ln1b + l * EE, xn);
        gemm_wmma<<<gE, blk, 0, stream>>>(xn, Wq + we, bq + l * EE, nullptr, nullptr, qh, MM, EE, EE, MODE_QKV);
        gemm_wmma<<<gE, blk, 0, stream>>>(xn, Wk + we, bk + l * EE, nullptr, nullptr, kh, MM, EE, EE, MODE_QKV);
        gemm_wmma<<<gE, blk, 0, stream>>>(xn, Wv + we, bv + l * EE, nullptr, nullptr, vh, MM, EE, EE, MODE_QKV);
        attn_kernel<<<(BBATCH * NH * (TT / 16)) / 8, blk, 0, stream>>>(qh, kh, vh, yh);
        gemm_wmma<<<gE, blk, 0, stream>>>(yh, Wp + we, bp + l * EE, h, h, nullptr, MM, EE, EE, MODE_RES);
        ln_kernel<<<MM, blk, 0, stream>>>(h, ln2g + l * EE, ln2b + l * EE, xn);
        gemm_wmma<<<gF, blk, 0, stream>>>(xn, Wf1 + wf, bf1 + l * FF, nullptr, nullptr, hid, MM, FF, EE, MODE_GELU);
        gemm_wmma<<<gE, blk, 0, stream>>>(hid, Wf2 + wf, bf2 + l * EE, h, h, nullptr, MM, EE, FF, MODE_RES);
    }

    ln_kernel<<<MM, blk, 0, stream>>>(h, lnfg, lnfb, xn);
    gemm_wmma<<<gV, blk, 0, stream>>>(xn, headw, nullptr, nullptr, (float*)d_out, nullptr, MM, VV, EE, MODE_F32);
}
